// GAT_12575664243204
// MI455X (gfx1250) — matrix operand order repaired and checked by reading, not compiled, not run on hardware
//
#include <hip/hip_runtime.h>
#include <hip/hip_bf16.h>

// ---------------------------------------------------------------------------
// Problem constants
// ---------------------------------------------------------------------------
#define BATCH 8
#define CCH   5
#define NNODE 512
#define FIN   768
#define HH    4
#define DHID  128
#define DOUT  64
#define HDH   (HH*DHID)   // 512
#define HDO   (HH*DOUT)   // 256
#define COUT  (CCH*DOUT)  // 320
#define NEG_SLOPE 0.2f

// ---------------------------------------------------------------------------
// WMMA types (CDNA5 gfx1250, wave32)
// ---------------------------------------------------------------------------
typedef __attribute__((ext_vector_type(16))) __bf16 v16bf;
typedef __attribute__((ext_vector_type(8)))  float  v8f;
typedef unsigned short ushort_t;

union Frag { v16bf v; uint4 q[2]; };
union Acc  { v8f v; float f[8]; };

__device__ __forceinline__ ushort_t f2bf(float x) {
  union { float f; unsigned u; } cv; cv.f = x;
  unsigned u = cv.u;
  unsigned r = u + 0x7fffu + ((u >> 16) & 1u);  // RNE
  return (ushort_t)(r >> 16);
}
__device__ __forceinline__ float bf2f(ushort_t h) {
  union { unsigned u; float f; } cv; cv.u = ((unsigned)h) << 16; return cv.f;
}

// ---------------------------------------------------------------------------
// Batched bf16 WMMA GEMM:  C[z](bf16) = act( A[z] (MxK,bf16) * B[z] + bias )
//  - A: row-major [m][k], k contiguous.
//  - TRANSB=false: B given pre-transposed Bt[n][k] (k contiguous, ldb = row
//    stride in k-units) -> pure vector staging.
//  - TRANSB=true : B given [k][n] (ldb = n-row stride); transposed on staging.
//  - z -> b=z/zdiv, h=z%zdiv; operand ptr += b*outer + h*inner (element units)
//  M fixed = 512 (grid.x*BM covers exactly); K multiple of 32.
// Block tile 128x128x32, 8 waves (2x4), wave = 64x32 = 4x2 WMMA tiles.
// Double-buffered LDS staging.
// ---------------------------------------------------------------------------
#define BM 128
#define BN 128
#define BK 32
#define LDS_S 40   // ushorts per row: 32 + 8 pad (80 B rows, 16B aligned)

template <bool TRANSB, bool NGUARD>
__global__ __launch_bounds__(256)
void wmma_gemm(const ushort_t* __restrict__ Ap, long a_so, long a_si, int lda,
               const ushort_t* __restrict__ Bp, long b_so, long b_si, int ldb,
               const float* __restrict__ bias, long bias_si,
               ushort_t* __restrict__ Cp, long c_so, long c_si, int ldc,
               int K, int Nout, int zdiv, int relu)
{
  __shared__ __align__(16) ushort_t sA[2][BM * LDS_S];
  __shared__ __align__(16) ushort_t sB[2][BN * LDS_S];

  const int z = blockIdx.z;
  const int b = z / zdiv;
  const int h = z % zdiv;

  const ushort_t* Amat = Ap + b * a_so + h * a_si;
  const ushort_t* Bmat = Bp + b * b_so + h * b_si;
  ushort_t*       Cmat = Cp + b * c_so + h * c_si;
  const float*    biash = bias ? (bias + h * bias_si) : nullptr;

  const int m0 = blockIdx.x * BM;
  const int n0 = blockIdx.y * BN;
  const int tid  = threadIdx.x;
  const int lane = tid & 31;
  const int wave = tid >> 5;
  const int wm = wave >> 2;     // 0..1 -> 64-row slab
  const int wn = wave & 3;      // 0..3 -> 32-col slab

  const int halfsel = lane >> 4;
  const int l16     = lane & 15;
  const int a_kbase = halfsel * 8;   // ISA 16-bit A 16x32 lane layout
  const int b_kbase = halfsel * 8;   // same k order as A (table order)

  Acc acc[4][2];
  #pragma unroll
  for (int i = 0; i < 4; ++i)
    #pragma unroll
    for (int j = 0; j < 2; ++j)
      #pragma unroll
      for (int r = 0; r < 8; ++r) acc[i][j].f[r] = 0.f;

  // ---- staging (pure uint4 copies on the hot path) ----
  auto stageA = [&](int buf, int k0, int kpref) {
    #pragma unroll
    for (int e = 0; e < 2; ++e) {           // 512 uint4 total, 2 per thread
      int linear = tid + e * 256;
      int row = linear >> 2;                // 4 uint4 per 32-half row
      int q   = linear & 3;
      const ushort_t* gp = Amat + (long)(m0 + row) * lda + k0 + q * 8;
      uint4 v = *(const uint4*)gp;
      *(uint4*)(&sA[buf][row * LDS_S + q * 8]) = v;
      if (e == 0 && kpref >= 0)
        __builtin_prefetch(gp + (kpref - k0), 0, 1);  // speculative, dropped if OOB
    }
  };

  auto stageB = [&](int buf, int k0, int kpref) {
    if constexpr (!TRANSB) {
      #pragma unroll
      for (int e = 0; e < 2; ++e) {
        int linear = tid + e * 256;
        int nn = linear >> 2;
        int q  = linear & 3;
        int gn = n0 + nn;
        uint4 v;
        if (!NGUARD || gn < Nout) {
          const ushort_t* gp = Bmat + (long)gn * ldb + k0 + q * 8;
          v = *(const uint4*)gp;
          if (e == 0 && kpref >= 0) __builtin_prefetch(gp + (kpref - k0), 0, 1);
        } else {
          v.x = v.y = v.z = v.w = 0u;
        }
        *(uint4*)(&sB[buf][nn * LDS_S + q * 8]) = v;
      }
    } else {
      // B is [k][n]; transpose into sB[n][k]. 8 n per uint4 load.
      #pragma unroll
      for (int e = 0; e < 2; ++e) {
        int linear = tid + e * 256;          // 512 units of (8n x 1k)
        int kk  = linear >> 4;               // 0..31
        int nn8 = (linear & 15) * 8;
        int gn  = n0 + nn8;
        if (!NGUARD || gn + 7 < Nout) {
          const ushort_t* gp = Bmat + (long)(k0 + kk) * ldb + gn;
          uint4 v = *(const uint4*)gp;
          if (e == 0 && kpref >= 0) __builtin_prefetch(gp + (long)(kpref - k0) * ldb, 0, 1);
          const ushort_t* hv = (const ushort_t*)&v;
          #pragma unroll
          for (int t = 0; t < 8; ++t) sB[buf][(nn8 + t) * LDS_S + kk] = hv[t];
        } else {
          #pragma unroll
          for (int t = 0; t < 8; ++t) sB[buf][(nn8 + t) * LDS_S + kk] = 0;
        }
      }
    }
  };

  auto compute = [&](int buf) {
    Frag bfr[2];
    #pragma unroll
    for (int nt = 0; nt < 2; ++nt) {
      const ushort_t* bp = &sB[buf][(wn * 32 + nt * 16 + l16) * LDS_S + b_kbase];
      bfr[nt].q[0] = *(const uint4*)(bp);
      bfr[nt].q[1] = *(const uint4*)(bp + 16);
    }
    Frag af[4];
    #pragma unroll
    for (int mt = 0; mt < 4; ++mt) {
      const ushort_t* ap = &sA[buf][(wm * 64 + mt * 16 + l16) * LDS_S + a_kbase];
      af[mt].q[0] = *(const uint4*)(ap);
      af[mt].q[1] = *(const uint4*)(ap + 16);
    }
    #pragma unroll
    for (int mt = 0; mt < 4; ++mt)
      #pragma unroll
      for (int nt = 0; nt < 2; ++nt)
        acc[mt][nt].v = __builtin_amdgcn_wmma_f32_16x16x32_bf16(
            false, af[mt].v, false, bfr[nt].v,
            (short)0, acc[mt][nt].v, false, false);
  };

  // ---- software-pipelined main loop ----
  const int ktiles = K / BK;
  stageA(0, 0, BK);
  stageB(0, 0, BK);
  __syncthreads();
  int cur = 0;
  for (int kt = 0; kt < ktiles; ++kt) {
    if (kt + 1 < ktiles) {
      int knext = (kt + 1) * BK;
      int kpref = (kt + 2 < ktiles) ? (kt + 2) * BK : -1;
      stageA(cur ^ 1, knext, kpref);
      stageB(cur ^ 1, knext, kpref);
    }
    compute(cur);
    __syncthreads();
    cur ^= 1;
  }

  // ---- epilogue: bias + optional ReLU -> bf16 ----
  #pragma unroll
  for (int mt = 0; mt < 4; ++mt)
    #pragma unroll
    for (int nt = 0; nt < 2; ++nt)
      #pragma unroll
      for (int r = 0; r < 8; ++r) {
        int gr = m0 + wm * 64 + mt * 16 + halfsel * 8 + r;
        int gc = n0 + wn * 32 + nt * 16 + l16;
        if (!NGUARD || gc < Nout) {
          float v = acc[mt][nt].f[r];
          if (biash) v += biash[gc];
          if (relu)  v = fmaxf(v, 0.f);
          Cmat[(long)gr * ldc + gc] = f2bf(v);
        }
      }
}

// ---------------------------------------------------------------------------
// One-time converters
// ---------------------------------------------------------------------------
// x[:,0] (B,N,FIN) fp32 -> bf16
__global__ void cvt_x_kernel(const float* __restrict__ x, ushort_t* __restrict__ dst)
{
  long t = (long)blockIdx.x * blockDim.x + threadIdx.x;
  const long per = (long)NNODE * FIN;
  if (t >= (long)BATCH * per) return;
  long b = t / per, f = t % per;
  dst[t] = f2bf(x[b * (long)CCH * per + f]);
}

// W (C,K,N) fp32 -> Wt (C,N,K) bf16   (grid.z = C)
__global__ void cvt_wt_kernel(const float* __restrict__ W, ushort_t* __restrict__ Wt,
                              int K, int N)
{
  int t = blockIdx.x * blockDim.x + threadIdx.x;
  if (t >= K * N) return;
  int n = t % N, k = t / N;   // coalesced read along n
  const long off = (long)blockIdx.z * K * N;
  Wt[off + (long)n * K + k] = f2bf(W[off + (long)k * N + n]);
}

// ---------------------------------------------------------------------------
// el/er projections from bf16 features
// ---------------------------------------------------------------------------
__global__ void elr_kernel(const ushort_t* __restrict__ feat, int ldf,
                           const float* __restrict__ al, const float* __restrict__ ar,
                           float* __restrict__ el, float* __restrict__ er,
                           int D, int Hh, int total /*B*N*/)
{
  int t = blockIdx.x * blockDim.x + threadIdx.x;
  if (t >= total * Hh) return;
  int h  = t % Hh;
  int bn = t / Hh;
  const ushort_t* fr = feat + (long)bn * ldf + h * D;
  const float* alh = al + h * D;
  const float* arh = ar + h * D;
  float sl = 0.f, sr = 0.f;
  for (int d = 0; d < D; ++d) {
    float f = bf2f(fr[d]);
    sl += f * alh[d];
    sr += f * arh[d];
  }
  el[t] = sl;
  er[t] = sr;
}

// ---------------------------------------------------------------------------
// Masked softmax over source nodes i per (b,h,j); writes bf16 alpha^T
// ---------------------------------------------------------------------------
__global__ __launch_bounds__(256)
void attn_kernel(const float* __restrict__ adj, const float* __restrict__ el,
                 const float* __restrict__ er, ushort_t* __restrict__ alphaT,
                 int c, int Cc, int Nn, int Hh)
{
  __shared__ float red[256];
  __shared__ float ecache[NNODE];
  const int j = blockIdx.x, h = blockIdx.y, b = blockIdx.z;
  const int tid = threadIdx.x;

  const float erj = er[((long)b * Nn + j) * Hh + h];
  const float* adjcol = adj + (((long)b * Cc + c) * Nn) * (long)Nn + j;

  float mloc = -INFINITY;
  for (int i = tid; i < Nn; i += 256) {
    float aval = adjcol[(long)i * Nn];
    float e;
    if (aval > 0.f) {
      float xv = el[((long)b * Nn + i) * Hh + h] + erj;
      e = (xv > 0.f) ? xv : NEG_SLOPE * xv;
    } else {
      e = -INFINITY;
    }
    ecache[i] = e;
    mloc = fmaxf(mloc, e);
  }
  red[tid] = mloc;
  __syncthreads();
  for (int s = 128; s > 0; s >>= 1) {
    if (tid < s) red[tid] = fmaxf(red[tid], red[tid + s]);
    __syncthreads();
  }
  const float mx = red[0];
  __syncthreads();

  float sloc = 0.f;
  for (int i = tid; i < Nn; i += 256) {
    float e = ecache[i];
    float p = (e == -INFINITY) ? 0.f : __expf(e - mx);
    ecache[i] = p;
    sloc += p;
  }
  red[tid] = sloc;
  __syncthreads();
  for (int s = 128; s > 0; s >>= 1) {
    if (tid < s) red[tid] += red[tid + s];
    __syncthreads();
  }
  const float inv = 1.f / fmaxf(red[0], 1e-9f);

  ushort_t* arow = alphaT + ((long)(b * Hh + h) * Nn + j) * (long)Nn;
  for (int i = tid; i < Nn; i += 256) arow[i] = f2bf(ecache[i] * inv);
}

// ---------------------------------------------------------------------------
// Head-mean into output slice c (fp32 out)
// ---------------------------------------------------------------------------
__global__ void mean_kernel(const ushort_t* __restrict__ hin, float* __restrict__ out,
                            int c, int total /*B*N*/, int Hh, int Dd, int Cw)
{
  int t = blockIdx.x * blockDim.x + threadIdx.x;
  if (t >= total * Dd) return;
  int d  = t % Dd;
  int bn = t / Dd;
  float s = 0.f;
  for (int h = 0; h < Hh; ++h) s += bf2f(hin[(long)bn * (Hh * Dd) + h * Dd + d]);
  out[(long)bn * Cw + c * Dd + d] = s * (1.0f / Hh);
}

// ---------------------------------------------------------------------------
// Host launch
// ---------------------------------------------------------------------------
extern "C" void kernel_launch(void* const* d_in, const int* in_sizes, int n_in,
                              void* d_out, int out_size, void* d_ws, size_t ws_size,
                              hipStream_t stream) {
  (void)in_sizes; (void)n_in; (void)out_size; (void)ws_size;

  const float* x   = (const float*)d_in[0];
  const float* adj = (const float*)d_in[1];
  const float* W0  = (const float*)d_in[2];
  const float* al0 = (const float*)d_in[3];
  const float* ar0 = (const float*)d_in[4];
  const float* b0  = (const float*)d_in[5];
  const float* W1  = (const float*)d_in[6];
  const float* al1 = (const float*)d_in[7];
  const float* ar1 = (const float*)d_in[8];
  const float* b1  = (const float*)d_in[9];
  const float* W2  = (const float*)d_in[10];
  const float* al2 = (const float*)d_in[11];
  const float* ar2 = (const float*)d_in[12];
  const float* b2  = (const float*)d_in[13];
  float* out = (float*)d_out;

  // ---- workspace layout (all bf16 except el/er) ----
  char* ws = (char*)d_ws;
  size_t off = 0;
  ushort_t* hb     = (ushort_t*)(ws + off); off += (size_t)BATCH * NNODE * FIN * 2;   // 6.29 MB
  ushort_t* featb  = (ushort_t*)(ws + off); off += (size_t)BATCH * NNODE * HDH * 2;   // 4.19 MB
  ushort_t* hbufb  = (ushort_t*)(ws + off); off += (size_t)BATCH * NNODE * HDH * 2;   // 4.19 MB
  ushort_t* alphaT = (ushort_t*)(ws + off); off += (size_t)BATCH * HH * NNODE * NNODE * 2; // 16.8 MB
  float*    el     = (float*)(ws + off);    off += (size_t)BATCH * NNODE * HH * 4;
  float*    er     = (float*)(ws + off);    off += (size_t)BATCH * NNODE * HH * 4;
  ushort_t* Wt0    = (ushort_t*)(ws + off); off += (size_t)CCH * FIN * HDH * 2;       // 3.93 MB
  ushort_t* Wt1    = (ushort_t*)(ws + off); off += (size_t)CCH * HDH * HDH * 2;       // 2.62 MB
  ushort_t* Wt2    = (ushort_t*)(ws + off); off += (size_t)CCH * HDH * HDO * 2;       // 1.31 MB

  const dim3 blk(256);
  const long NN = (long)NNODE * NNODE;

  // ---- one-time conversions (per launch) ----
  cvt_x_kernel<<<(int)(((long)BATCH * NNODE * FIN + 255) / 256), blk, 0, stream>>>(x, hb);
  cvt_wt_kernel<<<dim3((FIN * HDH + 255) / 256, 1, CCH), blk, 0, stream>>>(W0, Wt0, FIN, HDH);
  cvt_wt_kernel<<<dim3((HDH * HDH + 255) / 256, 1, CCH), blk, 0, stream>>>(W1, Wt1, HDH, HDH);
  cvt_wt_kernel<<<dim3((HDH * HDO + 255) / 256, 1, CCH), blk, 0, stream>>>(W2, Wt2, HDH, HDO);

  for (int c = 0; c < CCH; ++c) {
    // ================= layer 0 =================
    // feat = h0 @ W0[c]  (M=512,K=768,N=512), B pre-transposed
    wmma_gemm<false, false><<<dim3(NNODE / BM, HDH / BN, BATCH), blk, 0, stream>>>(
        hb, (long)NNODE * FIN, 0, FIN,
        Wt0 + (long)c * FIN * HDH, 0, 0, FIN,
        nullptr, 0,
        featb, (long)NNODE * HDH, 0, HDH,
        FIN, HDH, 1, 0);
    elr_kernel<<<(BATCH * NNODE * HH + 255) / 256, blk, 0, stream>>>(
        featb, HDH, al0 + c * HH * DHID, ar0 + c * HH * DHID, el, er, DHID, HH, BATCH * NNODE);
    attn_kernel<<<dim3(NNODE, HH, BATCH), blk, 0, stream>>>(adj, el, er, alphaT, c, CCH, NNODE, HH);
    // hbuf = relu(alpha^T @ feat + b0)  per (b,h): M=512,K=512,N=128
    wmma_gemm<true, false><<<dim3(NNODE / BM, 1, BATCH * HH), blk, 0, stream>>>(
        alphaT, (long)HH * NN, NN, NNODE,
        featb, (long)NNODE * HDH, DHID, HDH,
        b0 + (long)c * HDH, DHID,
        hbufb, (long)NNODE * HDH, DHID, HDH,
        NNODE, DHID, HH, 1);

    // ================= layer 1 =================
    wmma_gemm<false, false><<<dim3(NNODE / BM, HDH / BN, BATCH), blk, 0, stream>>>(
        hbufb, (long)NNODE * HDH, 0, HDH,
        Wt1 + (long)c * HDH * HDH, 0, 0, HDH,
        nullptr, 0,
        featb, (long)NNODE * HDH, 0, HDH,
        HDH, HDH, 1, 0);
    elr_kernel<<<(BATCH * NNODE * HH + 255) / 256, blk, 0, stream>>>(
        featb, HDH, al1 + c * HH * DHID, ar1 + c * HH * DHID, el, er, DHID, HH, BATCH * NNODE);
    attn_kernel<<<dim3(NNODE, HH, BATCH), blk, 0, stream>>>(adj, el, er, alphaT, c, CCH, NNODE, HH);
    wmma_gemm<true, false><<<dim3(NNODE / BM, 1, BATCH * HH), blk, 0, stream>>>(
        alphaT, (long)HH * NN, NN, NNODE,
        featb, (long)NNODE * HDH, DHID, HDH,
        b1 + (long)c * HDH, DHID,
        hbufb, (long)NNODE * HDH, DHID, HDH,
        NNODE, DHID, HH, 1);

    // ================= layer 2 =================
    // feat2 = h @ W2[c]  (M=512,K=512,N=256)
    wmma_gemm<false, false><<<dim3(NNODE / BM, HDO / BN, BATCH), blk, 0, stream>>>(
        hbufb, (long)NNODE * HDH, 0, HDH,
        Wt2 + (long)c * HDH * HDO, 0, 0, HDH,
        nullptr, 0,
        featb, (long)NNODE * HDO, 0, HDO,
        HDH, HDO, 1, 0);
    elr_kernel<<<(BATCH * NNODE * HH + 255) / 256, blk, 0, stream>>>(
        featb, HDO, al2 + c * HH * DOUT, ar2 + c * HH * DOUT, el, er, DOUT, HH, BATCH * NNODE);
    attn_kernel<<<dim3(NNODE, HH, BATCH), blk, 0, stream>>>(adj, el, er, alphaT, c, CCH, NNODE, HH);
    // out2 = alpha^T @ feat2 + b2 (no relu), N=64 -> column guards
    wmma_gemm<true, true><<<dim3(NNODE / BM, 1, BATCH * HH), blk, 0, stream>>>(
        alphaT, (long)HH * NN, NN, NNODE,
        featb, (long)NNODE * HDO, DOUT, HDO,
        b2 + (long)c * HDO, DOUT,
        hbufb, (long)NNODE * HDO, DOUT, HDO,
        NNODE, DOUT, HH, 0);

    // mean over heads -> out[:, :, c*64:(c+1)*64]
    mean_kernel<<<(BATCH * NNODE * DOUT + 255) / 256, blk, 0, stream>>>(
        hbufb, out, c, BATCH * NNODE, HH, DOUT, COUT);
  }
}
